// MultiDQNGNN_74586402062945
// MI455X (gfx1250) — matrix operand order repaired and checked by reading, not compiled, not run on hardware
//
#include <hip/hip_runtime.h>
#include <hip/hip_bf16.h>

typedef __attribute__((ext_vector_type(16))) _Float16 v16h;
typedef __attribute__((ext_vector_type(8)))  float    v8f;

#define BN_EPS 1e-5f

struct ConvParams {
  const float* w[4];
  const float* b[4];
  const float* g[4];
  const float* bt[4];
  const float* mn[4];
  const float* vr[4];
};

__device__ inline v8f wmma16x16x32(v16h a, v16h b, v8f c) {
  // D = A(16x32 f16) * B(32x16 f16) + C(16x16 f32)
  return __builtin_amdgcn_wmma_f32_16x16x32_f16(false, a, false, b, (short)0, c,
                                                false, false);
}

// ---------------------------------------------------------------------------
// Kernel 1: full 4-layer conv+BN+ReLU stack for one image, resident in LDS.
// Input  : state[img] = [16][30][30] f32 (NCHW)
// Output : X1[img]    = [7744] f16, flattened as c*484 + h*22 + w (ref order)
// ---------------------------------------------------------------------------
__global__ __launch_bounds__(256) void conv_stack_kernel(
    const float* __restrict__ state, _Float16* __restrict__ X1, ConvParams P) {
  extern __shared__ char smem[];
  _Float16* buf0 = (_Float16*)smem;       // 21632 halves
  _Float16* buf1 = buf0 + 21632;          // 12544 halves
  _Float16* wst  = buf1 + 12544;          // 4608 halves
  float* sc = (float*)(wst + 4608);       // 32 floats
  float* bo = sc + 32;                    // 32 floats

  const int t = threadIdx.x;
  const int lane = t & 31;
  const int wave = t >> 5;
  const int img = blockIdx.x;

  // Stage input [16][30][30] f32 -> channel-last f16 buf0[(h*30+w)*16 + c]
  const float* gin = state + (size_t)img * 14400;
  for (int idx = t; idx < 14400; idx += 256) {
    int c = idx / 900, r = idx % 900;
    buf0[r * 16 + c] = (_Float16)gin[idx];
  }

  const int CinA[4]  = {16, 16, 32, 16};
  const int CoutA[4] = {16, 32, 16, 16};
  const int HinA[4]  = {30, 28, 26, 24};

#pragma unroll
  for (int layer = 0; layer < 4; ++layer) {
    const int cin  = CinA[layer];
    const int cout = CoutA[layer];
    const int win  = HinA[layer];
    const int wout = win - 2;
    const int Mtot = wout * wout;
    const int Ktot = cin * 9;
    const int kSteps = (Ktot + 31) / 32;
    _Float16* inB  = (layer & 1) ? buf1 : buf0;
    _Float16* outB = (layer & 1) ? buf0 : buf1;

    // Fold BN (eval) into per-channel scale/bias
    if (t < cout) {
      float s = P.g[layer][t] * rsqrtf(P.vr[layer][t] + BN_EPS);
      sc[t] = s;
      bo[t] = (P.b[layer][t] - P.mn[layer][t]) * s + P.bt[layer][t];
    }
    __syncthreads();

    for (int nt = 0; nt < cout / 16; ++nt) {
      // Stage B tile: wst[ks][n][k]; K=ks*32+k -> (shift s=K/cin, chan ci=K%cin)
      const float* wg = P.w[layer];
      for (int idx = t; idx < kSteps * 512; idx += 256) {
        int ks = idx >> 9, r = idx & 511;
        int n = r >> 5, k = r & 31;
        int K = ks * 32 + k;
        int s = K / cin, ci = K % cin;
        int co = nt * 16 + n;
        _Float16 v = (_Float16)0.0f;
        if (s < 9) v = (_Float16)(wg[(co * cin + ci) * 9 + s] * sc[co]);
        wst[idx] = v;
      }
      __syncthreads();

      const int nTiles = (Mtot + 15) / 16;
      for (int tile = wave; tile < nTiles; tile += 8) {
        int p = tile * 16 + (lane & 15);          // A row = output pixel
        if (p >= Mtot) p = Mtot - 1;              // clamp; store is guarded
        int y = p / wout, x = p % wout;
        int chofs = (lane >> 4) * 8;              // A-layout half-wave K skew
        v8f acc = {};
        for (int ks = 0; ks < kSteps; ++ks) {
          v16h a;
#pragma unroll
          for (int g = 0; g < 2; ++g) {
            // halves [g*8 .. g*8+7] hold K = ks*32 + g*16 + chofs + 0..7
            int Kb = ks * 32 + g * 16 + chofs;
            int s = Kb / cin, ci = Kb % cin;      // uniform across lanes
            if (s < 9) {
              int dy = s / 3, dx = s % 3;
              const _Float16* src = inB + ((y + dy) * win + (x + dx)) * cin + ci;
#pragma unroll
              for (int i = 0; i < 8; ++i) a[g * 8 + i] = src[i];
            } else {
#pragma unroll
              for (int i = 0; i < 8; ++i) a[g * 8 + i] = (_Float16)0.0f;
            }
          }
          v16h bfr;
          {
            const _Float16* src =
                wst + ks * 512 + (lane & 15) * 32 + (lane >> 4) * 8;
#pragma unroll
            for (int i = 0; i < 8; ++i) { bfr[i] = src[i]; bfr[8 + i] = src[16 + i]; }
          }
          acc = wmma16x16x32(a, bfr, acc);
        }
        // C layout: VGPR r -> M = r + 8*(lane>=16), N = lane&15
        int n = lane & 15;
        int co = nt * 16 + n;
        int mb = (lane >> 4) * 8;
        int pbase = tile * 16 + mb;
        if (tile * 16 + 16 <= Mtot) {             // full tile: uniform branch
#pragma unroll
          for (int r = 0; r < 8; ++r) {
            float v = fmaxf(acc[r] + bo[co], 0.0f);
            outB[(pbase + r) * cout + co] = (_Float16)v;
          }
        } else {
#pragma unroll
          for (int r = 0; r < 8; ++r) {
            int p2 = pbase + r;
            if (p2 < Mtot) {
              float v = fmaxf(acc[r] + bo[co], 0.0f);
              outB[p2 * cout + co] = (_Float16)v;
            }
          }
        }
      }
      __syncthreads();
    }
  }

  // conv4 output is buf0 channel-last [p][16]; write in ref flatten order.
  _Float16* go = X1 + (size_t)img * 7744;
  for (int idx = t; idx < 7744; idx += 256) {
    int c = idx / 484, r = idx % 484;             // idx = c*484 + h*22 + w
    go[idx] = buf0[r * 16 + c];
  }
}

// ---------------------------------------------------------------------------
// Kernel 2: generic WMMA GEMM  C[M,N] = act( A[M,K](f16) * B[K,N] + bias )
// Block = 128 threads (4 waves), block tile 64x16, K step 64 (2 wmma/round).
// grid = (ceil(M/64), N/16, batch); sA/sB/sC are batch strides (elements).
// ---------------------------------------------------------------------------
template <typename TB>
__global__ __launch_bounds__(128) void gemm_ws_kernel(
    const _Float16* __restrict__ A, int lda, long long sA,
    const TB* __restrict__ B, int ldb, long long sB,
    const float* __restrict__ bias, _Float16* __restrict__ C, int ldc,
    long long sC, int M, int N, int K, int relu) {
  __shared__ _Float16 As[64 * 72];   // 64 rows, 64 halves + pad
  __shared__ _Float16 Bs[16 * 64];   // [n][k] (B transposed)
  const int t = threadIdx.x;
  const int lane = t & 31, wave = t >> 5;
  const int m0 = blockIdx.x * 64;
  const int n0 = blockIdx.y * 16;
  const int z = blockIdx.z;
  const _Float16* Ab = A + (size_t)z * sA;
  const TB* Bb = B + (size_t)z * sB;
  _Float16* Cb = C + (size_t)z * sC;

  const int arow = t >> 1, acb = (t & 1) * 32;   // A-staging assignment
  v8f acc = {};
  for (int kk = 0; kk < K; kk += 64) {
    {  // cooperative A load: 64x64 halves
      int m = m0 + arow;
#pragma unroll
      for (int i = 0; i < 32; ++i) {
        int k = kk + acb + i;
        _Float16 v = (_Float16)0.0f;
        if (m < M && k < K) v = Ab[(size_t)m * lda + k];
        As[arow * 72 + acb + i] = v;
      }
      // Prefetch next A tile into cache (global_prefetch_b8 path)
      if (m < M && kk + 64 < K)
        __builtin_prefetch(&Ab[(size_t)m * lda + kk + 64 + acb], 0, 3);
    }
    {  // cooperative B load: 64x16 -> Bs[n][k] with f16 convert
#pragma unroll
      for (int j = 0; j < 8; ++j) {
        int idx = t * 8 + j;
        int n = idx >> 6, k = idx & 63;
        int kg = kk + k;
        _Float16 v = (_Float16)0.0f;
        if (kg < K) v = (_Float16)(float)Bb[(size_t)kg * ldb + n0 + n];
        Bs[n * 64 + k] = v;
      }
    }
    __syncthreads();
#pragma unroll
    for (int ks = 0; ks < 2; ++ks) {
      v16h a, b;
      const _Float16* s = As + (wave * 16 + (lane & 15)) * 72 + ks * 32;
      int chofs = (lane >> 4) * 8;
#pragma unroll
      for (int i = 0; i < 8; ++i) {
        a[i] = s[chofs + i];
        a[8 + i] = s[16 + chofs + i];
      }
      const _Float16* sb = Bs + (lane & 15) * 64 + ks * 32 + (lane >> 4) * 8;
#pragma unroll
      for (int i = 0; i < 8; ++i) { b[i] = sb[i]; b[8 + i] = sb[16 + i]; }
      acc = wmma16x16x32(a, b, acc);
    }
    __syncthreads();
  }
  // Epilogue: C layout VGPR r -> M = r + 8*(lane>=16), N = lane&15
  int n = lane & 15;
  int col = n0 + n;
  float bv = bias ? bias[col] : 0.0f;
  int mbase = m0 + wave * 16 + (lane >> 4) * 8;
  if (m0 + wave * 16 + 16 <= M) {                 // full tile: uniform branch
#pragma unroll
    for (int r = 0; r < 8; ++r) {
      float v = acc[r] + bv;
      if (relu) v = fmaxf(v, 0.0f);
      Cb[(size_t)(mbase + r) * ldc + col] = (_Float16)v;
    }
  } else {
#pragma unroll
    for (int r = 0; r < 8; ++r) {
      int m = mbase + r;
      if (m < M) {
        float v = acc[r] + bv;
        if (relu) v = fmaxf(v, 0.0f);
        Cb[(size_t)m * ldc + col] = (_Float16)v;
      }
    }
  }
}

// ---------------------------------------------------------------------------
// Kernel 3: GCN symmetric normalization; writes norm^T (64x64) as f16.
// ---------------------------------------------------------------------------
__global__ __launch_bounds__(64) void gcn_norm_kernel(
    const float* __restrict__ adj, _Float16* __restrict__ normT) {
  __shared__ float ah[64 * 64];
  __shared__ float dinv[64];
  const int t = threadIdx.x;
  for (int idx = t; idx < 4096; idx += 64) {
    int i = idx >> 6, j = idx & 63;
    float v = adj[idx];
    if (i == j && v == 0.0f) v = 1.0f;   // add_remaining_self_loops
    ah[idx] = v;
  }
  __syncthreads();
  float d = 0.0f;                        // deg[j] = sum_i a_hat[i][j]
  for (int i = 0; i < 64; ++i) d += ah[i * 64 + t];
  dinv[t] = (d > 0.0f) ? rsqrtf(d) : 0.0f;
  __syncthreads();
  for (int idx = t; idx < 4096; idx += 64) {
    int i = idx >> 6, j = idx & 63;
    normT[j * 64 + i] = (_Float16)(dinv[i] * ah[idx] * dinv[j]);
  }
}

// ---------------------------------------------------------------------------
// Kernel 4: final head layer (128 -> 5) + inactive-agent mask, f32 out.
// ---------------------------------------------------------------------------
__global__ __launch_bounds__(256) void head3_mask_kernel(
    const _Float16* __restrict__ H, const float* __restrict__ W,
    const float* __restrict__ b, const int* __restrict__ inactives,
    float* __restrict__ out) {
  int row = blockIdx.x * blockDim.x + threadIdx.x;  // row = batch*64 + agent
  if (row >= 2048) return;
  int agent = row & 63;
  bool inact = inactives[agent] != 0;
  float acc[5];
#pragma unroll
  for (int a = 0; a < 5; ++a) acc[a] = b[a];
  for (int k = 0; k < 128; ++k) {
    float x = (float)H[row * 128 + k];
#pragma unroll
    for (int a = 0; a < 5; ++a) acc[a] += x * W[k * 5 + a];
  }
#pragma unroll
  for (int a = 0; a < 5; ++a) out[row * 5 + a] = inact ? 0.0f : acc[a];
}

// ---------------------------------------------------------------------------
extern "C" void kernel_launch(void* const* d_in, const int* in_sizes, int n_in,
                              void* d_out, int out_size, void* d_ws,
                              size_t ws_size, hipStream_t stream) {
  (void)in_sizes; (void)n_in; (void)out_size; (void)ws_size;
  const float* state = (const float*)d_in[0];
  const float* adj   = (const float*)d_in[1];
  const int*   inact = (const int*)d_in[2];

  ConvParams cp;
  for (int i = 0; i < 4; ++i) {
    cp.w[i]  = (const float*)d_in[3 + 6 * i + 0];
    cp.b[i]  = (const float*)d_in[3 + 6 * i + 1];
    cp.g[i]  = (const float*)d_in[3 + 6 * i + 2];
    cp.bt[i] = (const float*)d_in[3 + 6 * i + 3];
    cp.mn[i] = (const float*)d_in[3 + 6 * i + 4];
    cp.vr[i] = (const float*)d_in[3 + 6 * i + 5];
  }
  const float *mlpw[3], *mlpb[3];
  for (int j = 0; j < 3; ++j) {
    mlpw[j] = (const float*)d_in[27 + 2 * j];
    mlpb[j] = (const float*)d_in[28 + 2 * j];
  }
  const float* gcnw = (const float*)d_in[33];
  const float* gcnb = (const float*)d_in[34];
  const float *hw[3], *hb[3];
  for (int j = 0; j < 3; ++j) {
    hw[j] = (const float*)d_in[35 + 2 * j];
    hb[j] = (const float*)d_in[36 + 2 * j];
  }

  // Workspace partition (f16, 256B aligned)
  char* ws = (char*)d_ws;
  size_t cur = 0;
  auto allocH = [&](size_t nelem) {
    _Float16* p = (_Float16*)(ws + cur);
    cur += (nelem * sizeof(_Float16) + 255) & ~(size_t)255;
    return p;
  };
  const int BN = 2048, E = 128, FLAT = 7744, NA = 64;
  _Float16* X1 = allocH((size_t)BN * FLAT);  // conv features
  _Float16* X2 = allocH((size_t)BN * E);     // mlp1 out
  _Float16* X3 = allocH((size_t)BN * E);     // mlp2 out
  _Float16* FE = allocH((size_t)BN * E);     // encoder feats
  _Float16* XW = allocH((size_t)BN * E);     // feats @ gcn.w
  _Float16* NT = allocH((size_t)NA * NA);    // norm^T
  _Float16* EM = allocH((size_t)BN * E);     // gcn embeddings
  _Float16* H1 = allocH((size_t)BN * E);
  _Float16* H2 = allocH((size_t)BN * E);

  // 1. Conv stack, LDS-resident per image.
  size_t convSh = (21632 + 12544 + 4608) * sizeof(_Float16) + 64 * sizeof(float);
  conv_stack_kernel<<<dim3(2048), dim3(256), convSh, stream>>>(state, X1, cp);

  // 2. Encoder MLP: 7744->128 (relu), 128->128 (relu), 128->128.
  gemm_ws_kernel<float><<<dim3(32, 8, 1), 128, 0, stream>>>(
      X1, FLAT, 0, mlpw[0], E, 0, mlpb[0], X2, E, 0, BN, E, FLAT, 1);
  gemm_ws_kernel<float><<<dim3(32, 8, 1), 128, 0, stream>>>(
      X2, E, 0, mlpw[1], E, 0, mlpb[1], X3, E, 0, BN, E, E, 1);
  gemm_ws_kernel<float><<<dim3(32, 8, 1), 128, 0, stream>>>(
      X3, E, 0, mlpw[2], E, 0, mlpb[2], FE, E, 0, BN, E, E, 0);

  // 3. GCN: norm matrix, xw = feats @ W, emb[b] = norm^T @ xw[b] + bias.
  gcn_norm_kernel<<<dim3(1), dim3(64), 0, stream>>>(adj, NT);
  gemm_ws_kernel<float><<<dim3(32, 8, 1), 128, 0, stream>>>(
      FE, E, 0, gcnw, E, 0, (const float*)nullptr, XW, E, 0, BN, E, E, 0);
  gemm_ws_kernel<_Float16><<<dim3(1, 8, 32), 128, 0, stream>>>(
      NT, NA, 0, XW, E, (long long)NA * E, gcnb, EM, E, (long long)NA * E,
      NA, E, NA, 0);

  // 4. DQN head: 128->128 (relu) x2, then 128->5 + mask.
  gemm_ws_kernel<float><<<dim3(32, 8, 1), 128, 0, stream>>>(
      EM, E, 0, hw[0], E, 0, hb[0], H1, E, 0, BN, E, E, 1);
  gemm_ws_kernel<float><<<dim3(32, 8, 1), 128, 0, stream>>>(
      H1, E, 0, hw[1], E, 0, hb[1], H2, E, 0, BN, E, E, 1);
  head3_mask_kernel<<<dim3(8), dim3(256), 0, stream>>>(H2, hw[2], hb[2], inact,
                                                       (float*)d_out);
}
